// GIN_18382460027174
// MI455X (gfx1250) — matrix operand order repaired and checked by reading, not compiled, not run on hardware
//
#include <hip/hip_runtime.h>

// GIN forward, MI455X (gfx1250, wave32, WMMA).
//
// Roofline: GEMMs total ~9.8 GFLOP (negligible at WMMA rates); node buffers
// (25.6 MB) fit in 192 MB L2; runtime is dominated by the edge scatter-add
// (3 x 1.6M x 256B gathers + ~307M f32 L2 atomics). GEMMs use f16 WMMA with
// f32 accumulation; BatchNorm folded into per-channel scale/bias.

#define NGRAPHS_ 256
#define C_       64     // IN_C == HID
#define MID_     128    // 2*HID
#define OUTC_    16
#define BN_EPS_  1e-5f

typedef _Float16 half_t;
typedef __attribute__((ext_vector_type(16))) _Float16 v16h;
typedef __attribute__((ext_vector_type(8)))  float    v8f;

// ---------------------------------------------------------------- utilities
__global__ void zero_f32(float* __restrict__ p, int n) {
  int i = blockIdx.x * blockDim.x + threadIdx.x;
  int stride = gridDim.x * blockDim.x;
  for (; i < n; i += stride) p[i] = 0.0f;
}

// ------------------------------------------------------- edge scatter (agg)
// 16 lanes per edge, float4 per lane: coalesced 64B chunks of the 256B row.
__global__ void scatter_edges(const float* __restrict__ x,
                              const int* __restrict__ src,
                              const int* __restrict__ dst,
                              float* __restrict__ agg, int nE) {
  int t = blockIdx.x * blockDim.x + threadIdx.x;
  int e = t >> 4;
  if (e >= nE) return;
  int c = (t & 15) << 2;
  // look-ahead prefetch of a future source row (emits global_prefetch_b8)
  int ep = e + 64;
  if (ep < nE) __builtin_prefetch(x + (size_t)src[ep] * C_ + c, 0, 0);
  int s = src[e], d = dst[e];
  float4 v = *(const float4*)(x + (size_t)s * C_ + c);
  float* o = agg + (size_t)d * C_ + c;
  atomicAdd(o + 0, v.x);
  atomicAdd(o + 1, v.y);
  atomicAdd(o + 2, v.z);
  atomicAdd(o + 3, v.w);
}

// ------------------------------------------------- fused GIN MLP via WMMA
// Block = 128 threads = 4 waves; each wave owns a 16-node tile.
//   h = (1+eps)*x + agg                       (f32 -> f16, LDS)
//   mid = relu(bn_mid(h @ W1 + b1))           (WMMA f16 -> f32 acc, LDS f16)
//   out = relu(bn_out(mid @ W2 + b2))         (WMMA f16 -> f32 acc, global)
// BN+bias folded: y -> s*y + t per channel.

#define LDH_   68    // sH row stride (halves), 8B-aligned float4 stores
#define LDK1_  72    // sW1t row stride (k up to 64)
#define LDMID_ 136   // sMid / sW2t row stride (k up to 128)

// A-fragment, 16-bit A 16x32 (contiguous k, not the ISA 7.12.2 table order):
// lane<16: K=0..15 ; lane>=16: K=16..31 (contiguous, matches B).
__device__ inline v16h load_A_frag(const half_t* __restrict__ tile, int ldt,
                                   int kb, int m, int hi) {
  const half_t* row = tile + m * ldt + kb * 32;
  v16h a;
#pragma unroll
  for (int v = 0; v < 8; ++v) {
    int k = 16 * hi + 2 * v;
    a[2 * v]     = row[k];
    a[2 * v + 1] = row[k + 1];
  }
  return a;
}

// B-fragment, 16-bit B 32x16: lanes 0-15 hold K=0..15 (N=lane), 16-31 K=16..31.
// Weights staged transposed in LDS: wt[n*ldk + k] = W[k][n].
__device__ inline v16h load_B_frag(const half_t* __restrict__ wt, int ldk,
                                   int n, int kb, int hi) {
  const half_t* col = wt + n * ldk + kb * 32 + 16 * hi;
  v16h b;
#pragma unroll
  for (int i = 0; i < 16; ++i) b[i] = col[i];
  return b;
}

__global__ __launch_bounds__(128) void gin_mlp(
    const float* __restrict__ xin, const float* __restrict__ agg,
    const float* __restrict__ epsp,
    const float* __restrict__ W1, const float* __restrict__ b1,
    const float* __restrict__ g1, const float* __restrict__ bb1,
    const float* __restrict__ m1, const float* __restrict__ v1,
    const float* __restrict__ W2, const float* __restrict__ b2,
    const float* __restrict__ g2, const float* __restrict__ bb2,
    const float* __restrict__ m2, const float* __restrict__ v2,
    float* __restrict__ xout, int nN) {
  __shared__ half_t sW1t[MID_ * LDK1_];      // W1^T : [128][72]
  __shared__ half_t sW2t[C_ * LDMID_];       // W2^T : [64][136]
  __shared__ float  sS1[MID_], sT1[MID_], sS2[C_], sT2[C_];
  __shared__ half_t sH[4][16 * LDH_];        // per-wave h tile (f16)
  __shared__ half_t sMid[4][16 * LDMID_];    // per-wave mid tile (f16)

  int tid = threadIdx.x;
  // stage transposed f16 weights
  for (int i = tid; i < C_ * MID_; i += 128) {      // W1: [64][128]
    int k = i >> 7, n = i & 127;
    sW1t[n * LDK1_ + k] = (half_t)W1[i];
  }
  for (int i = tid; i < MID_ * C_; i += 128) {      // W2: [128][64]
    int k = i >> 6, n = i & 63;
    sW2t[n * LDMID_ + k] = (half_t)W2[i];
  }
  if (tid < MID_) {
    float s = g1[tid] * rsqrtf(v1[tid] + BN_EPS_);
    sS1[tid] = s;
    sT1[tid] = s * (b1[tid] - m1[tid]) + bb1[tid];
  }
  if (tid < C_) {
    float s = g2[tid] * rsqrtf(v2[tid] + BN_EPS_);
    sS2[tid] = s;
    sT2[tid] = s * (b2[tid] - m2[tid]) + bb2[tid];
  }
  __syncthreads();

  int wave = tid >> 5, lane = tid & 31;
  int m = lane & 15, hi = lane >> 4;
  int base = blockIdx.x * 64 + wave * 16;
  float eps1 = 1.0f + epsp[0];

  // build h tile in LDS (coalesced float4 reads; zero-fill OOB rows)
  half_t* ht = sH[wave];
  for (int u = lane; u < 16 * 16; u += 32) {   // 256 float4 units
    int r = u >> 4, c4 = (u & 15) << 2;
    int node = base + r;
    float4 hv = make_float4(0.f, 0.f, 0.f, 0.f);
    if (node < nN) {
      float4 xv = *(const float4*)(xin + (size_t)node * C_ + c4);
      float4 av = *(const float4*)(agg + (size_t)node * C_ + c4);
      hv.x = eps1 * xv.x + av.x;
      hv.y = eps1 * xv.y + av.y;
      hv.z = eps1 * xv.z + av.z;
      hv.w = eps1 * xv.w + av.w;
    }
    half_t* p = ht + r * LDH_ + c4;
    p[0] = (half_t)hv.x; p[1] = (half_t)hv.y;
    p[2] = (half_t)hv.z; p[3] = (half_t)hv.w;
  }

  // GEMM1: [16x64] @ [64x128], K in 2 blocks of 32, N in 8 tiles of 16
  v16h A1[2];
#pragma unroll
  for (int kb = 0; kb < 2; ++kb) A1[kb] = load_A_frag(ht, LDH_, kb, m, hi);

  half_t* mt = sMid[wave];
#pragma unroll
  for (int n0 = 0; n0 < 8; ++n0) {
    v8f acc = {};
#pragma unroll
    for (int kb = 0; kb < 2; ++kb) {
      v16h b = load_B_frag(sW1t, LDK1_, n0 * 16 + m, kb, hi);
      acc = __builtin_amdgcn_wmma_f32_16x16x32_f16(
          false, A1[kb], false, b, (short)0, acc, false, false);
    }
    int ch = n0 * 16 + m;
    float s = sS1[ch], t = sT1[ch];
#pragma unroll
    for (int r = 0; r < 8; ++r) {   // D row M = r + 8*hi, col N = ch
      float val = fmaxf(fmaf(s, acc[r], t), 0.0f);
      mt[(r + 8 * hi) * LDMID_ + ch] = (half_t)val;
    }
  }

  // GEMM2: [16x128] @ [128x64], K in 4 blocks, N in 4 tiles
  v16h A2[4];
#pragma unroll
  for (int kb = 0; kb < 4; ++kb) A2[kb] = load_A_frag(mt, LDMID_, kb, m, hi);

#pragma unroll
  for (int n0 = 0; n0 < 4; ++n0) {
    v8f acc = {};
#pragma unroll
    for (int kb = 0; kb < 4; ++kb) {
      v16h b = load_B_frag(sW2t, LDMID_, n0 * 16 + m, kb, hi);
      acc = __builtin_amdgcn_wmma_f32_16x16x32_f16(
          false, A2[kb], false, b, (short)0, acc, false, false);
    }
    int ch = n0 * 16 + m;
    float s = sS2[ch], t = sT2[ch];
#pragma unroll
    for (int r = 0; r < 8; ++r) {
      int node = base + r + 8 * hi;
      if (node < nN)
        xout[(size_t)node * C_ + ch] = fmaxf(fmaf(s, acc[r], t), 0.0f);
    }
  }
}

// ------------------------------------------------------------ global pool
__global__ void pool_add(const float* __restrict__ x,
                         const int* __restrict__ batch,
                         float* __restrict__ pooled, int nN) {
  int t = blockIdx.x * blockDim.x + threadIdx.x;
  int node = t >> 4;
  if (node >= nN) return;
  int c = (t & 15) << 2;
  int g = batch[node];
  float4 v = *(const float4*)(x + (size_t)node * C_ + c);
  float* o = pooled + (size_t)g * C_ + c;
  atomicAdd(o + 0, v.x);
  atomicAdd(o + 1, v.y);
  atomicAdd(o + 2, v.z);
  atomicAdd(o + 3, v.w);
}

// ------------------------------------------------------------------ head
// 256 threads = 256 graphs; 2.6 MFLOP total -> scalar f32 is free, exact.
__global__ __launch_bounds__(256) void head_kernel(
    const float* __restrict__ pooled,
    const float* __restrict__ Wh1, const float* __restrict__ bh1,
    const float* __restrict__ gg, const float* __restrict__ bb,
    const float* __restrict__ mm, const float* __restrict__ vv,
    const float* __restrict__ Wh2, const float* __restrict__ bh2,
    float* __restrict__ out) {
  __shared__ float sW[C_ * C_];
  __shared__ float sSc[C_], sTb[C_];
  __shared__ float sW2[C_ * OUTC_];
  __shared__ float sB2[OUTC_];
  int tid = threadIdx.x;
  for (int i = tid; i < C_ * C_; i += 256) sW[i] = Wh1[i];
  for (int i = tid; i < C_ * OUTC_; i += 256) sW2[i] = Wh2[i];
  if (tid < C_) {
    float s = gg[tid] * rsqrtf(vv[tid] + BN_EPS_);
    sSc[tid] = s;
    sTb[tid] = s * (bh1[tid] - mm[tid]) + bb[tid];
  }
  if (tid < OUTC_) sB2[tid] = bh2[tid];
  __syncthreads();

  float hid[C_];
#pragma unroll
  for (int c = 0; c < C_; ++c) hid[c] = 0.0f;
  for (int k = 0; k < C_; ++k) {
    float gk = pooled[(size_t)tid * C_ + k];
#pragma unroll
    for (int c = 0; c < C_; ++c) hid[c] = fmaf(gk, sW[k * C_ + c], hid[c]);
  }
#pragma unroll
  for (int c = 0; c < C_; ++c) hid[c] = fmaxf(fmaf(sSc[c], hid[c], sTb[c]), 0.0f);
#pragma unroll
  for (int o = 0; o < OUTC_; ++o) {
    float acc = sB2[o];
    for (int c = 0; c < C_; ++c) acc = fmaf(hid[c], sW2[c * OUTC_ + o], acc);
    out[(size_t)tid * OUTC_ + o] = acc;
  }
}

// -------------------------------------------------------------- launcher
// d_in order (setup_inputs dict order; params flattened as jax pytree,
// i.e. dict keys sorted at every level):
//   0 x [N,64] f32 | 1 edge_index [2,E] i32 | 2 batch [N] i32
//   3..6   head.bn1 {b,g,m,v}
//   7,8    head.lin1 {W,b} | 9,10 head.lin2 {W,b}
//   11+13l layer l: bn_mid{b,g,m,v}, bn_out{b,g,m,v}, eps, lin1{W,b}, lin2{W,b}
extern "C" void kernel_launch(void* const* d_in, const int* in_sizes, int n_in,
                              void* d_out, int out_size, void* d_ws,
                              size_t ws_size, hipStream_t stream) {
  const float* x     = (const float*)d_in[0];
  const int*   ei    = (const int*)d_in[1];
  const int*   batch = (const int*)d_in[2];
  int nN = in_sizes[2];
  int nE = in_sizes[1] / 2;

  auto L = [&](int l, int k) { return (const float*)d_in[11 + 13 * l + k]; };
  // k: 0..3 bn_mid{b,g,m,v}, 4..7 bn_out{b,g,m,v}, 8 eps,
  //    9 lin1.W, 10 lin1.b, 11 lin2.W, 12 lin2.b

  float* ws = (float*)d_ws;
  size_t n64 = (size_t)nN * C_;
  float* bufA   = ws;              // x after layer 0 / 2
  float* bufB   = bufA + n64;      // x after layer 1
  float* aggb   = bufB + n64;      // neighbor sums
  float* pooled = aggb + n64;      // [256,64]

  const int* src = ei;
  const int* dst = ei + nE;

  for (int l = 0; l < 3; ++l) {
    const float* xin = (l == 0) ? x : (l == 1 ? bufA : bufB);
    float* xout      = (l == 0) ? bufA : (l == 1 ? bufB : bufA);
    zero_f32<<<2048, 256, 0, stream>>>(aggb, (int)n64);
    scatter_edges<<<(nE * 16 + 255) / 256, 256, 0, stream>>>(xin, src, dst,
                                                             aggb, nE);
    gin_mlp<<<(nN + 63) / 64, 128, 0, stream>>>(
        xin, aggb, L(l, 8),
        L(l, 9), L(l, 10), L(l, 1), L(l, 0), L(l, 2), L(l, 3),
        L(l, 11), L(l, 12), L(l, 5), L(l, 4), L(l, 6), L(l, 7),
        xout, nN);
  }

  zero_f32<<<64, 256, 0, stream>>>(pooled, NGRAPHS_ * C_);
  pool_add<<<(nN * 16 + 255) / 256, 256, 0, stream>>>(bufA, batch, pooled, nN);
  head_kernel<<<1, 256, 0, stream>>>(
      pooled,
      (const float*)d_in[7], (const float*)d_in[8],
      (const float*)d_in[4], (const float*)d_in[3],
      (const float*)d_in[5], (const float*)d_in[6],
      (const float*)d_in[9], (const float*)d_in[10],
      (float*)d_out);
}
